// SelfAttentionHead_13924283974050
// MI455X (gfx1250) — matrix operand order repaired and checked by reading, not compiled, not run on hardware
//
#include <hip/hip_runtime.h>
#include <hip/hip_bf16.h>

// ---------------------------------------------------------------------------
// Causal single-head attention for MI455X (gfx1250, wave32, WMMA bf16).
//   B=8, T=4096, E=1024, H=128, fp32 in/out.
// Phase 1: QKV projection GEMM (fp32 -> bf16 via v_cvt_pk_bf16_f32,
//          v_wmma_f32_16x16x32_bf16). V is stored TRANSPOSED (B,H,T).
// Phase 2: flash-attention, S^T = K Q^T and O^T = V^T P^T formulation.
//          K and V^T tiles are staged into LDS by the Tensor Data Mover with
//          DOUBLE BUFFERING: tensor_load_to_lds for tile i+2 is issued while
//          tile i is consumed; s_wait_tensorcnt 2 overlaps DMA with WMMA.
// ---------------------------------------------------------------------------

typedef __attribute__((ext_vector_type(16))) __bf16 v16bf;
typedef __attribute__((ext_vector_type(8)))  float  v8f;
typedef __attribute__((ext_vector_type(4)))  unsigned int u32x4;
typedef __attribute__((ext_vector_type(8)))  int i32x8;
typedef __attribute__((ext_vector_type(4)))  int i32x4;

struct __align__(16) U4 { unsigned int x, y, z, w; };
struct __align__(16) F4 { float a, b, c, d; };

union BF16Frag {
  v16bf v;
  unsigned int u[8];
};

__device__ __forceinline__ unsigned short f2bf_sw(float f) {
  unsigned int u = __float_as_uint(f);
  u += 0x7fffu + ((u >> 16) & 1u);   // round-to-nearest-even
  return (unsigned short)(u >> 16);
}
// Packed f32x2 -> bf16x2: builtin if present, else the gfx1250 ISA op via
// inline asm, else software (host pass).
__device__ __forceinline__ unsigned int cvtpk2bf(float lo, float hi) {
#if defined(__gfx1250__) && __has_builtin(__builtin_amdgcn_cvt_pk_bf16_f32)
  auto r = __builtin_amdgcn_cvt_pk_bf16_f32(lo, hi);
  static_assert(sizeof(r) == 4, "unexpected cvt_pk_bf16 result size");
  return __builtin_bit_cast(unsigned int, r);
#elif defined(__gfx1250__)
  unsigned int r;
  asm("v_cvt_pk_bf16_f32 %0, %1, %2" : "=v"(r) : "v"(lo), "v"(hi));
  return r;
#else
  return (unsigned int)f2bf_sw(lo) | ((unsigned int)f2bf_sw(hi) << 16);
#endif
}
__device__ __forceinline__ unsigned short f2bf(float f) {
#if defined(__gfx1250__)
  return (unsigned short)(cvtpk2bf(f, 0.0f) & 0xffffu);
#else
  return f2bf_sw(f);
#endif
}
__device__ __forceinline__ void setfrag(BF16Frag& f, U4 a, U4 b) {
  f.u[0] = a.x; f.u[1] = a.y; f.u[2] = a.z; f.u[3] = a.w;
  f.u[4] = b.x; f.u[5] = b.y; f.u[6] = b.z; f.u[7] = b.w;
}

constexpr int Bb = 8;
constexpr int Tt = 4096;
constexpr int Ee = 1024;
constexpr int Hh = 128;

#if defined(__gfx1250__) && __has_builtin(__builtin_amdgcn_tensor_load_to_lds)
#define USE_TDM 1
#else
#define USE_TDM 0
#endif

constexpr int KT_ELEMS = 32 * 136;    // K tile in LDS (pad 8 bf16/row)
constexpr int VT_ELEMS = 128 * 40;    // V^T tile in LDS (pad 8 bf16/row)
constexpr int NBUF     = USE_TDM ? 2 : 1;

#if USE_TDM
// 2D tile load global->LDS via Tensor Data Mover (bf16 elements).
// pad_interval_code: pad after 2^(code+1) DWORDs; pad_amount_code: (code+1) DWORDs.
// 6-arg builtin form (clang-23 / therock-10.0): groups 2/3 + extra group zero.
__device__ __forceinline__ void tdm_load_2d(unsigned lds_off,
                                            unsigned long long gaddr,
                                            unsigned tensor_d0, unsigned tensor_d1,
                                            unsigned tile_d0, unsigned tile_d1,
                                            unsigned long long d0_stride,
                                            unsigned pad_interval_code,
                                            unsigned pad_amount_code) {
  u32x4 g0;
  g0[0] = 1u;                                   // count=1, user descriptor
  g0[1] = lds_off;                              // LDS byte address
  g0[2] = (unsigned)(gaddr & 0xffffffffull);    // global addr [31:0]
  g0[3] = (unsigned)((gaddr >> 32) & 0x01ffffffull) | (2u << 30);  // [56:32] | type=2

  i32x8 g1;
  g1[0] = (int)((1u << 16)                      // data_size = 2 bytes
              | (1u << 20)                      // pad_enable
              | (pad_interval_code << 22)
              | (pad_amount_code << 25));
  g1[1] = (int)((tensor_d0 & 0xffffu) << 16);                 // atomic_addr=0 | dim0 lo
  g1[2] = (int)(((tensor_d0 >> 16) & 0xffffu) | ((tensor_d1 & 0xffffu) << 16));
  g1[3] = (int)(((tensor_d1 >> 16) & 0xffffu) | ((tile_d0 & 0xffffu) << 16));
  g1[4] = (int)(tile_d1 & 0xffffu);                           // tile_dim1 | tile_dim2=0
  g1[5] = (int)(d0_stride & 0xffffffffull);                   // dim0_stride [31:0]
  g1[6] = (int)((d0_stride >> 32) & 0xffffull);               // dim0_stride [47:32]
  g1[7] = 0;

  i32x4 z4 = {0, 0, 0, 0};
  i32x8 z8 = {0, 0, 0, 0, 0, 0, 0, 0};
  __builtin_amdgcn_tensor_load_to_lds(g0, g1, z4, z4, z8, 0);
}
#endif

// ---------------------------------------------------------------------------
// Phase 1: out[m, 0..127] = x[m, :] @ W  (bf16 result), M = B*T = 32768.
// Workgroup = 8 waves, each wave owns a 16-row stripe, full N=128.
// blockIdx.y selects Wq / Wk / Wv; V output is stored transposed (B,H,T).
// ---------------------------------------------------------------------------
__global__ __launch_bounds__(256)
void qkv_gemm_kernel(const float* __restrict__ x,
                     const float* __restrict__ Wq,
                     const float* __restrict__ Wk,
                     const float* __restrict__ Wv,
                     unsigned short* __restrict__ Qo,
                     unsigned short* __restrict__ Ko,
                     unsigned short* __restrict__ Vto)
{
  __shared__ __align__(16) unsigned short wT[128 * 40];   // 10240 B

  const int tid  = threadIdx.x;
  const int wave = tid >> 5;
  const int lane = tid & 31;
  const int ml   = lane & 15;
  const int hi   = lane >> 4;

  const int sel = blockIdx.y;
  const float* W = (sel == 0) ? Wq : (sel == 1) ? Wk : Wv;

  const int m0 = blockIdx.x * 128 + wave * 16;
  const float* xrow = x + (size_t)(m0 + ml) * Ee;

  v8f acc[8];
#pragma unroll
  for (int i = 0; i < 8; ++i)
#pragma unroll
    for (int r = 0; r < 8; ++r) acc[i][r] = 0.0f;

  for (int k0 = 0; k0 < Ee; k0 += 32) {
    __syncthreads();
    // Stage W^T chunk: pairs (k, k+1) for one n -> one packed cvt + b32 store.
#pragma unroll
    for (int i = 0; i < 8; ++i) {
      int g  = tid + i * 256;        // 0..2047 pair index
      int kl = (g >> 7) * 2;         // even k within chunk
      int n  = g & 127;
      float w0 = W[(size_t)(k0 + kl) * Hh + n];
      float w1 = W[(size_t)(k0 + kl + 1) * Hh + n];
      *(unsigned int*)&wT[n * 40 + kl] = cvtpk2bf(w0, w1);
    }
    __syncthreads();

    // A-fragment: x rows (16m x 32k), A layout: k = 16*hi + 2*v  (16 consecutive k).
    BF16Frag a;
    const float* p0 = xrow + k0 + hi * 16;
    const float* p1 = p0 + 8;
#pragma unroll
    for (int j = 0; j < 4; ++j) {
      a.u[j]     = cvtpk2bf(p0[2 * j], p0[2 * j + 1]);
      a.u[4 + j] = cvtpk2bf(p1[2 * j], p1[2 * j + 1]);
    }

    // B-fragments: W^T rows as B layout: k = 16*hi + 2*v  (16 consecutive k).
#pragma unroll
    for (int nc = 0; nc < 8; ++nc) {
      BF16Frag b;
      const unsigned short* wp = &wT[(nc * 16 + ml) * 40 + hi * 16];
      U4 b0 = *(const U4*)(wp);
      U4 b1 = *(const U4*)(wp + 8);
      setfrag(b, b0, b1);
      acc[nc] = __builtin_amdgcn_wmma_f32_16x16x32_bf16(
          false, a.v, false, b.v, (short)0, acc[nc], false, false);
    }
  }

  // C layout: row m = m0 + r + 8*hi, col = nc*16 + ml.
  if (sel == 2) {
    // V stored transposed: Vt[b][h][t]. Per lane the 8 rows are 8 consecutive
    // t for a fixed h -> one packed b128 store per chunk.
    const int bb = m0 >> 12;               // m0 / T
    const int tb = (m0 & (Tt - 1)) + 8 * hi;
    unsigned short* vtg = Vto + (size_t)bb * Hh * Tt;
#pragma unroll
    for (int nc = 0; nc < 8; ++nc) {
      const int col = nc * 16 + ml;        // h
      U4 d;
      d.x = cvtpk2bf(acc[nc][0], acc[nc][1]);
      d.y = cvtpk2bf(acc[nc][2], acc[nc][3]);
      d.z = cvtpk2bf(acc[nc][4], acc[nc][5]);
      d.w = cvtpk2bf(acc[nc][6], acc[nc][7]);
      *(U4*)(vtg + (size_t)col * Tt + tb) = d;
    }
  } else {
    unsigned short* out = (sel == 0) ? Qo : Ko;
#pragma unroll
    for (int nc = 0; nc < 8; ++nc) {
      const int col = nc * 16 + ml;
#pragma unroll
      for (int r = 0; r < 8; ++r) {
        out[(size_t)(m0 + 8 * hi + r) * Hh + col] = f2bf(acc[nc][r]);
      }
    }
  }
}

// ---------------------------------------------------------------------------
// Phase 2: flash attention. Workgroup = 8 waves x 16 queries = 128 queries.
// Double-buffered TDM staging; per 32-key step: 8 WMMAs for S^T, 8 for O^T.
// ---------------------------------------------------------------------------
__global__ __launch_bounds__(256)
void attn_kernel(const unsigned short* __restrict__ Q,
                 const unsigned short* __restrict__ K,
                 const unsigned short* __restrict__ Vt,
                 float* __restrict__ out)
{
  __shared__ __align__(16) unsigned short kt[NBUF * KT_ELEMS];
  __shared__ __align__(16) unsigned short vt[NBUF * VT_ELEMS];

  const int tid  = threadIdx.x;
  const int wave = tid >> 5;
  const int lane = tid & 31;
  const int ml   = lane & 15;
  const int hi   = lane >> 4;

  const int b    = blockIdx.y;
  const int qblk = blockIdx.x * 128;
  const int q0   = qblk + wave * 16;       // wave's query base
  const int mg   = q0 + ml;                // this lane's query (column of S^T)

  const unsigned short* Qb  = Q  + (size_t)b * Tt * Hh;
  const unsigned short* Kb  = K  + (size_t)b * Tt * Hh;
  const unsigned short* Vtb = Vt + (size_t)b * Hh * Tt;

  // Q^T B-fragments (persistent): B layout k = (v<4?0:16) + 8*hi + 2*(v%4) within 32-h chunk.
  BF16Frag qf[4];
  {
    const unsigned short* qr = Qb + (size_t)mg * Hh;
#pragma unroll
    for (int c = 0; c < 4; ++c) {
      U4 a0 = *(const U4*)(qr + c * 32 + hi * 8);
      U4 a1 = *(const U4*)(qr + c * 32 + hi * 8 + 16);
      setfrag(qf[c], a0, a1);
    }
  }

  v8f acc[8];                               // O^T accumulators (128h x 16m)
#pragma unroll
  for (int i = 0; i < 8; ++i)
#pragma unroll
    for (int r = 0; r < 8; ++r) acc[i][r] = 0.0f;

  float run_max = -1e30f;
  float lsum    = 0.0f;
  const float scale = 0.08838834764831845f;  // 1/sqrt(128)

  const int sEnd = qblk + 128;               // exclusive key bound for this block

#if USE_TDM
  // K tile: 32 rows x 128 bf16, pad 4 DW/row -> LDS stride 136 elems.
  // V^T tile: 128 rows x 32 bf16 (global stride T), pad 4 DW/row -> stride 40.
  const unsigned long long kgbase = (unsigned long long)(size_t)Kb;
  const unsigned long long vgbase = (unsigned long long)(size_t)Vtb;
  auto issue_tile = [&](int s, int buf) {
    tdm_load_2d((unsigned)(size_t)&kt[buf * KT_ELEMS],
                kgbase + (unsigned long long)s * Hh * 2,
                Hh, (unsigned)Tt, Hh, 32,
                (unsigned long long)Hh, 5u /*64 DW*/, 3u /*4 DW*/);
    tdm_load_2d((unsigned)(size_t)&vt[buf * VT_ELEMS],
                vgbase + (unsigned long long)s * 2,
                (unsigned)Tt, Hh, 32, Hh,
                (unsigned long long)Tt, 3u /*16 DW*/, 3u /*4 DW*/);
  };
  if (wave == 0) {
    issue_tile(0, 0);
    if (32 < sEnd) issue_tile(32, 1);
  }
#endif

  int it = 0;
  for (int s0 = 0; s0 < sEnd; s0 += 32, ++it) {
    const unsigned short* ktp;
    const unsigned short* vtp;
#if USE_TDM
    const int buf = it & 1;
    if (wave == 0) {
      // Tile it complete (TENSORcnt retires in order); tile it+1 may be in flight.
      if (s0 + 32 < sEnd) {
        __builtin_amdgcn_s_wait_tensorcnt(2);
      } else {
        __builtin_amdgcn_s_wait_tensorcnt(0);
      }
    }
    __syncthreads();                       // tile it visible to all waves
    ktp = kt + buf * KT_ELEMS;
    vtp = vt + buf * VT_ELEMS;
#else
    __syncthreads();                       // previous tile fully consumed
#pragma unroll
    for (int i = 0; i < 2; ++i) {
      int e   = (tid + i * 256) * 8;       // K: element index in 32x128 tile
      int row = e >> 7;
      int col = e & 127;
      U4 dk = *(const U4*)(Kb + (size_t)(s0 + row) * Hh + col);
      *(U4*)&kt[row * 136 + col] = dk;
      int ev  = tid + i * 256;             // V^T: segment index in 128x32 tile
      int h   = ev >> 2;
      int seg = (ev & 3) * 8;
      U4 dv = *(const U4*)(Vtb + (size_t)h * Tt + s0 + seg);
      *(U4*)&vt[h * 40 + seg] = dv;
    }
    __syncthreads();
    ktp = kt;
    vtp = vt;
#endif

    // Wave-uniform causal activity: any key in this tile visible to this wave?
    const bool active = (s0 <= q0 + 15);
    if (active) {
      // S^T tiles: rows s (two 16-row tiles), cols m.  A = K rows, B = Q^T.
      v8f st0, st1;
#pragma unroll
      for (int r = 0; r < 8; ++r) { st0[r] = 0.0f; st1[r] = 0.0f; }
#pragma unroll
      for (int c = 0; c < 4; ++c) {
        const unsigned short* kp0 = &ktp[ml * 136 + c * 32 + hi * 8];
        const unsigned short* kp1 = &ktp[(16 + ml) * 136 + c * 32 + hi * 8];
        BF16Frag ka, kb2;
        setfrag(ka,  *(const U4*)kp0, *(const U4*)(kp0 + 16));
        setfrag(kb2, *(const U4*)kp1, *(const U4*)(kp1 + 16));
        st0 = __builtin_amdgcn_wmma_f32_16x16x32_bf16(
            false, ka.v, false, qf[c].v, (short)0, st0, false, false);
        st1 = __builtin_amdgcn_wmma_f32_16x16x32_bf16(
            false, kb2.v, false, qf[c].v, (short)0, st1, false, false);
      }

      // Scale + causal mask.  C layout: row s = s0 + tile*16 + 8*hi + r, col = ml.
      float t0[8], t1[8];
#pragma unroll
      for (int r = 0; r < 8; ++r) {
        int sg = s0 + 8 * hi + r;
        t0[r] = (sg      <= mg) ? st0[r] * scale : -1e30f;
        t1[r] = (sg + 16 <= mg) ? st1[r] * scale : -1e30f;
      }

      // Online softmax: per-column (= per-lane) max/sum; one lane^16 shuffle each.
      float vmax = t0[0];
#pragma unroll
      for (int r = 0; r < 8; ++r) {
        vmax = fmaxf(vmax, t0[r]);
        vmax = fmaxf(vmax, t1[r]);
      }
      vmax = fmaxf(vmax, __shfl_xor(vmax, 16, 32));
      float nmax = fmaxf(run_max, vmax);
      float corr = __expf(run_max - nmax);
      run_max = nmax;

      float psum = 0.0f;
#pragma unroll
      for (int r = 0; r < 8; ++r) {
        t0[r] = __expf(t0[r] - nmax);
        t1[r] = __expf(t1[r] - nmax);
        psum += t0[r] + t1[r];
      }
      psum += __shfl_xor(psum, 16, 32);
      lsum = lsum * corr + psum;

#pragma unroll
      for (int c2 = 0; c2 < 8; ++c2)
#pragma unroll
        for (int r = 0; r < 8; ++r) acc[c2][r] *= corr;

      // Repack P^T (C layout) into B-fragment (k = 16*hi + 2*v):
      // one cross-half exchange per register, then packed bf16 converts.
      float t0x[8], t1x[8];
#pragma unroll
      for (int r = 0; r < 8; ++r) {
        t0x[r] = __shfl_xor(t0[r], 16, 32);
        t1x[r] = __shfl_xor(t1[r], 16, 32);
      }
      BF16Frag pf;
#pragma unroll
      for (int v2 = 0; v2 < 4; ++v2) {
        float aLo = hi ? t1x[2 * v2]     : t0[2 * v2];
        float aHi = hi ? t1x[2 * v2 + 1] : t0[2 * v2 + 1];
        float bLo = hi ? t1[2 * v2]      : t0x[2 * v2];
        float bHi = hi ? t1[2 * v2 + 1]  : t0x[2 * v2 + 1];
        pf.u[v2]     = cvtpk2bf(aLo, aHi);
        pf.u[v2 + 4] = cvtpk2bf(bLo, bHi);
      }

      // O^T += V^T(chunk) @ P^T.  A = V^T rows h (A layout from LDS, stride 40).
#pragma unroll
      for (int c2 = 0; c2 < 8; ++c2) {
        const unsigned short* vp = &vtp[(c2 * 16 + ml) * 40 + hi * 16];
        BF16Frag va;
        setfrag(va, *(const U4*)vp, *(const U4*)(vp + 8));
        acc[c2] = __builtin_amdgcn_wmma_f32_16x16x32_bf16(
            false, va.v, false, pf.v, (short)0, acc[c2], false, false);
      }
    }

#if USE_TDM
    __syncthreads();                       // all waves done reading buf
    if (wave == 0 && s0 + 64 < sEnd) issue_tile(s0 + 64, buf);
#endif
  }

  // Epilogue: O(m, h) = O^T(h, m) / l.  Per lane: h = c2*16 + 8*hi + r, row = mg.
  const float inv = 1.0f / lsum;
  float* orow = out + ((size_t)b * Tt + mg) * Hh;
#pragma unroll
  for (int c2 = 0; c2 < 8; ++c2) {
    F4 lo  = {acc[c2][0] * inv, acc[c2][1] * inv, acc[c2][2] * inv, acc[c2][3] * inv};
    F4 hi4 = {acc[c2][4] * inv, acc[c2][5] * inv, acc[c2][6] * inv, acc[c2][7] * inv};
    *(F4*)(orow + c2 * 16 + 8 * hi)     = lo;
    *(F4*)(orow + c2 * 16 + 8 * hi + 4) = hi4;
  }
}

// ---------------------------------------------------------------------------
extern "C" void kernel_launch(void* const* d_in, const int* in_sizes, int n_in,
                              void* d_out, int out_size, void* d_ws, size_t ws_size,
                              hipStream_t stream) {
  const float* x  = (const float*)d_in[0];
  const float* Wq = (const float*)d_in[1];
  const float* Wk = (const float*)d_in[2];
  const float* Wv = (const float*)d_in[3];
  float* out = (float*)d_out;

  unsigned short* Qw = (unsigned short*)d_ws;          // bf16 Q (B,T,H)
  const size_t qs = (size_t)Bb * Tt * Hh;
  unsigned short* Kw  = Qw + qs;                       // bf16 K (B,T,H)
  unsigned short* Vtw = Kw + qs;                       // bf16 V^T (B,H,T)

  dim3 g1(Bb * Tt / 128, 3);
  qkv_gemm_kernel<<<g1, 256, 0, stream>>>(x, Wq, Wk, Wv, Qw, Kw, Vtw);

  dim3 g2(Tt / 128, Bb);
  attn_kernel<<<g2, 256, 0, stream>>>(Qw, Kw, Vtw, out);
}
